// FalconH1ParallelHybrid_14070312861954
// MI455X (gfx1250) — matrix operand order repaired and checked by reading, not compiled, not run on hardware
//
#include <hip/hip_runtime.h>
#include <hip/hip_bf16.h>
#include <math.h>

// ---------------- model dims ----------------
#define B_        2
#define L_        1024
#define HID_      2048
#define NH_       32
#define HD_       64
#define NKV_      8
#define INTER_    8192
#define DSSM_     2048
#define MH_       32
#define MP_       64
#define MS_       64
#define MK_       4
#define CONV_DIM_ (DSSM_ + 2 * MS_)          // 2176
#define PROJ_     (2 * DSSM_ + 2 * MS_ + MH_) // 4256
#define QKV_      (NH_ * HD_ + 2 * NKV_ * HD_) // 3072
#define M_        (B_ * L_)                   // 2048
#define EPS_      1e-5f
#define ROPE_BASE_ 1e11f
#define KEY_MULT_ 0.7f
#define ATTN_IN_  1.2f
#define ATTN_OUT_ 0.8f
#define SSM_IN_   1.1f
#define SSM_OUT_  0.9f
#define GATE_MULT_ 0.9f
#define DOWN_MULT_ 0.8f

// ---------------- types ----------------
typedef __attribute__((ext_vector_type(16))) __bf16        bf16x16;
typedef __attribute__((ext_vector_type(8)))  unsigned short u16x8;
typedef __attribute__((ext_vector_type(8)))  float          f32x8;

union Frag { bf16x16 bf; u16x8 u[2]; };

__device__ __forceinline__ unsigned short f32_to_bf16(float f) {
  union { float f; unsigned int u; } v; v.f = f;
  unsigned int u = v.u;
  u += 0x7FFFu + ((u >> 16) & 1u);  // round-to-nearest-even
  return (unsigned short)(u >> 16);
}

__device__ __forceinline__ float silu_f(float x) {
  return x / (1.0f + __expf(-x));
}

__device__ __forceinline__ float blockReduceSum256(float v, float* red) {
  int lane = threadIdx.x & 31, w = threadIdx.x >> 5;
#pragma unroll
  for (int off = 16; off; off >>= 1) v += __shfl_xor(v, off, 32);
  if (lane == 0) red[w] = v;
  __syncthreads();
  float t = (threadIdx.x < 8) ? red[threadIdx.x] : 0.0f;
#pragma unroll
  for (int off = 4; off; off >>= 1) t += __shfl_xor(t, off, 32);
  if (threadIdx.x == 0) red[0] = t;
  __syncthreads();
  return red[0];
}

// ---------------- TDM path probe (never launched; proves codegen) ----------------
#if __has_builtin(__builtin_amdgcn_tensor_load_to_lds)
typedef __attribute__((ext_vector_type(4))) unsigned int uint32x4_t_;
typedef __attribute__((ext_vector_type(8))) int          int32x8_t_;
typedef __attribute__((ext_vector_type(4))) int          int32x4_t_;
__global__ void tdm_probe_kernel(float* __restrict__ dst) {
  __shared__ float buf[64];
  uint32x4_t_ g0 = {};
  int32x8_t_  g1 = {};
  int32x4_t_  g2 = {};
  int32x4_t_  g3 = {};
#if defined(__clang_major__) && (__clang_major__ >= 23)
  int32x8_t_  g4 = {};
  __builtin_amdgcn_tensor_load_to_lds(g0, g1, g2, g3, g4, 0);
#else
  __builtin_amdgcn_tensor_load_to_lds(g0, g1, g2, g3, 0);
#endif
#if __has_builtin(__builtin_amdgcn_s_wait_tensorcnt)
  __builtin_amdgcn_s_wait_tensorcnt(0);
#endif
  __syncthreads();
  dst[threadIdx.x] = buf[threadIdx.x & 63];
}
#endif

// ---------------- cast fp32 -> bf16 ----------------
__global__ void cast_bf16_kernel(const float* __restrict__ src,
                                 unsigned short* __restrict__ dst, size_t n) {
  size_t i = (size_t)blockIdx.x * blockDim.x + threadIdx.x;
  size_t stride = (size_t)gridDim.x * blockDim.x;
  for (; i < n; i += stride) dst[i] = f32_to_bf16(src[i]);
}

// ---------------- input RMSNorm -> two bf16 activation copies ----------------
__global__ __launch_bounds__(256) void rmsnorm_in_kernel(
    const float* __restrict__ x, const float* __restrict__ w,
    unsigned short* __restrict__ hs_attn, unsigned short* __restrict__ hs_ssm) {
  __shared__ float red[8];
  int row = blockIdx.x;
  const float* xp = x + (size_t)row * HID_;
  float loc[HID_ / 256];
  float ss = 0.0f;
#pragma unroll
  for (int j = 0; j < HID_ / 256; ++j) {
    float v = xp[threadIdx.x + j * 256];
    loc[j] = v;
    ss += v * v;
  }
  float tot = blockReduceSum256(ss, red);
  float scale = rsqrtf(tot / HID_ + EPS_);
#pragma unroll
  for (int j = 0; j < HID_ / 256; ++j) {
    int c = threadIdx.x + j * 256;
    float hs = loc[j] * scale * w[c];
    hs_attn[(size_t)row * HID_ + c] = f32_to_bf16(hs * ATTN_IN_);
    hs_ssm[(size_t)row * HID_ + c]  = f32_to_bf16(hs * SSM_IN_);
  }
}

// ---------------- bf16 WMMA GEMM: C[MxN] = A[MxK] * W[NxK]^T ----------------
// 4 waves/block. Each wave: 32(M) x 64(N) tile, K-step 32, software-pipelined
// register double buffering so WMMAs overlap the next step's global loads.
__global__ __launch_bounds__(128) void gemm_bf16_wmma(
    const unsigned short* __restrict__ A, const unsigned short* __restrict__ W,
    float* __restrict__ C, int M, int N, int K) {
  int lane = threadIdx.x & 31;
  int wave = threadIdx.x >> 5;
  int m0 = blockIdx.y * 32;
  int n0 = blockIdx.x * 256 + wave * 64;
  if (n0 >= N || m0 >= M) return;

  int r = lane & 15;
  int hi = lane >> 4;
  int koffA = hi * 8;    // A frag: lanes 0-15 K{0..7,16..23}; lanes 16-31 K{8..15,24..31}
  int koffB = hi * 8;    // B frag: lanes 0-15 K{0..7,16..23}; lanes 16-31 K{8..15,24..31}

  const unsigned short* arow0 = A + (size_t)(m0 + r) * K + koffA;
  const unsigned short* arow1 = A + (size_t)(m0 + 16 + r) * K + koffA;
  const unsigned short* brow[4];
#pragma unroll
  for (int j = 0; j < 4; ++j) {
    int n = n0 + j * 16 + r;
    if (n > N - 1) n = N - 1;  // clamp for OOB tiles; stores are guarded below
    brow[j] = W + (size_t)n * K + koffB;
  }

  f32x8 acc[8] = {};

  // prologue: load k=0 fragments
  Frag a0c, a1c, bc[4];
  a0c.u[0] = *(const u16x8*)(arow0);
  a0c.u[1] = *(const u16x8*)(arow0 + 16);
  a1c.u[0] = *(const u16x8*)(arow1);
  a1c.u[1] = *(const u16x8*)(arow1 + 16);
#pragma unroll
  for (int j = 0; j < 4; ++j) {
    bc[j].u[0] = *(const u16x8*)(brow[j]);
    bc[j].u[1] = *(const u16x8*)(brow[j] + 16);
  }

  for (int k = 32; k < K; k += 32) {
    // issue next step's loads first ...
    Frag a0n, a1n, bn[4];
    a0n.u[0] = *(const u16x8*)(arow0 + k);
    a0n.u[1] = *(const u16x8*)(arow0 + k + 16);
    a1n.u[0] = *(const u16x8*)(arow1 + k);
    a1n.u[1] = *(const u16x8*)(arow1 + k + 16);
#pragma unroll
    for (int j = 0; j < 4; ++j) {
      bn[j].u[0] = *(const u16x8*)(brow[j] + k);
      bn[j].u[1] = *(const u16x8*)(brow[j] + k + 16);
    }
    __builtin_prefetch(arow0 + k + 256, 0, 1);
    // ... then compute on current fragments while loads are in flight
#pragma unroll
    for (int j = 0; j < 4; ++j)
      acc[j] = __builtin_amdgcn_wmma_f32_16x16x32_bf16(
          false, a0c.bf, false, bc[j].bf, (short)0, acc[j], false, false);
#pragma unroll
    for (int j = 0; j < 4; ++j)
      acc[4 + j] = __builtin_amdgcn_wmma_f32_16x16x32_bf16(
          false, a1c.bf, false, bc[j].bf, (short)0, acc[4 + j], false, false);
    a0c = a0n;
    a1c = a1n;
#pragma unroll
    for (int j = 0; j < 4; ++j) bc[j] = bn[j];
  }
  // epilogue compute
#pragma unroll
  for (int j = 0; j < 4; ++j)
    acc[j] = __builtin_amdgcn_wmma_f32_16x16x32_bf16(
        false, a0c.bf, false, bc[j].bf, (short)0, acc[j], false, false);
#pragma unroll
  for (int j = 0; j < 4; ++j)
    acc[4 + j] = __builtin_amdgcn_wmma_f32_16x16x32_bf16(
        false, a1c.bf, false, bc[j].bf, (short)0, acc[4 + j], false, false);

  // store: C/D layout -> lanes 0-15: M=i, lanes 16-31: M=8+i
  int madd = hi * 8;
#pragma unroll
  for (int t = 0; t < 2; ++t) {
#pragma unroll
    for (int j = 0; j < 4; ++j) {
      int n = n0 + j * 16 + r;
      if (n >= N) continue;
#pragma unroll
      for (int i = 0; i < 8; ++i) {
        int m = m0 + t * 16 + madd + i;
        C[(size_t)m * N + n] = acc[t * 4 + j][i];
      }
    }
  }
}

// ---------------- RoPE (in-place on qkv buffer), k also scaled by KEY_MULT ----------
__global__ void rope_kernel(float* __restrict__ qkv, const int* __restrict__ positions) {
  int idx = blockIdx.x * blockDim.x + threadIdx.x;
  int i = idx & 31;                 // frequency / pair index 0..31
  int t = idx >> 5;
  int head = t % (NH_ + NKV_);
  int row = t / (NH_ + NKV_);
  if (row >= M_) return;
  int l = row % L_;
  float pos = (float)positions[l];
  float inv = __expf(-(2.0f * (float)i / (float)HD_) * __logf(ROPE_BASE_));
  float ang = pos * inv;
  float s, c;
  __sincosf(ang, &s, &c);
  float mult = 1.0f;
  int col;
  if (head < NH_) {
    col = head * HD_;
  } else {
    col = NH_ * HD_ + (head - NH_) * HD_;
    mult = KEY_MULT_;
  }
  float* p = qkv + (size_t)row * QKV_ + col;
  float x1 = p[i] * mult, x2 = p[i + 32] * mult;
  p[i]      = x1 * c - x2 * s;
  p[i + 32] = x2 * c + x1 * s;
}

// ---------------- extract post-RoPE K (row-major bf16) and V^T (bf16) -------------
// k_bf:  [(b*NKV+g)*L + l]*HD + d      (contiguous in d)
// vT_bf: [((b*NKV+g)*HD + d)*L + l]    (contiguous in l)
__global__ void kv_extract_kernel(const float* __restrict__ qkv,
                                  unsigned short* __restrict__ k_bf,
                                  unsigned short* __restrict__ vT_bf) {
  size_t n = (size_t)M_ * NKV_ * HD_;
  size_t idx = (size_t)blockIdx.x * blockDim.x + threadIdx.x;
  if (idx >= n) return;
  int d = (int)(idx % HD_);
  int g = (int)((idx / HD_) % NKV_);
  int row = (int)(idx / (HD_ * NKV_));  // b*L + l
  int b = row / L_, l = row % L_;
  float kv = qkv[(size_t)row * QKV_ + NH_ * HD_ + g * HD_ + d];
  float vv = qkv[(size_t)row * QKV_ + NH_ * HD_ + NKV_ * HD_ + g * HD_ + d];
  k_bf[((size_t)(b * NKV_ + g) * L_ + l) * HD_ + d] = f32_to_bf16(kv);
  vT_bf[((size_t)(b * NKV_ + g) * HD_ + d) * L_ + l] = f32_to_bf16(vv);
}

// ---------------- WMMA flash attention ----------------
// One wave per (b, h, 16-query tile). Per 32-key step:
//   4 score WMMAs (QK^T over HD=64), causal mask + online softmax in C layout,
//   P transposed C->A layout through LDS (bf16), 4 PV WMMAs.
// Output written directly as bf16 for the o-projection GEMM.
__global__ __launch_bounds__(32) void attn_wmma_kernel(
    const float* __restrict__ qkv, const unsigned short* __restrict__ k_bf,
    const unsigned short* __restrict__ vT_bf, unsigned short* __restrict__ out_bf) {
  __shared__ __align__(16) unsigned short pbuf[16 * 32];

  int blk = blockIdx.x;
  int qt = blk % (L_ / 16);
  int h  = (blk / (L_ / 16)) % NH_;
  int b  = blk / ((L_ / 16) * NH_);
  int g  = h / (NH_ / NKV_);
  int q0 = qt * 16;

  int lane = threadIdx.x & 31;
  int r = lane & 15;
  int hi = lane >> 4;
  int koffA = hi * 8;
  int koffB = hi * 8;

  // ---- Q tile (16x64) -> two bf16 A fragments ----
  const float* qrow = qkv + ((size_t)b * L_ + q0 + r) * QKV_ + h * HD_;
  Frag qa[2];
#pragma unroll
  for (int f = 0; f < 2; ++f) {
#pragma unroll
    for (int e = 0; e < 8; ++e) {
      qa[f].u[0][e] = f32_to_bf16(qrow[f * 32 + koffA + e]);
      qa[f].u[1][e] = f32_to_bf16(qrow[f * 32 + koffA + 16 + e]);
    }
  }

  const unsigned short* kbase = k_bf + (size_t)(b * NKV_ + g) * L_ * HD_;
  const unsigned short* vbase = vT_bf + (size_t)(b * NKV_ + g) * HD_ * L_;

  f32x8 oacc[4] = {};
  float mrow[8], lrow[8];
#pragma unroll
  for (int i = 0; i < 8; ++i) { mrow[i] = -1e30f; lrow[i] = 0.0f; }

  int ntiles = (q0 + 47) / 32;  // cover keys 0 .. >= q0+15
  for (int tt = 0; tt < ntiles; ++tt) {
    int j0 = tt * 32;

    // ---- scores: two 16-key subtiles, 2 WMMAs each over HD=64 ----
    f32x8 S[2];
#pragma unroll
    for (int t = 0; t < 2; ++t) {
      const unsigned short* krow = kbase + (size_t)(j0 + 16 * t + r) * HD_;
      Frag kb0, kb1;
      kb0.u[0] = *(const u16x8*)(krow + koffB);
      kb0.u[1] = *(const u16x8*)(krow + koffB + 16);
      kb1.u[0] = *(const u16x8*)(krow + 32 + koffB);
      kb1.u[1] = *(const u16x8*)(krow + 32 + koffB + 16);
      f32x8 s = {};
      s = __builtin_amdgcn_wmma_f32_16x16x32_bf16(
          false, qa[0].bf, false, kb0.bf, (short)0, s, false, false);
      s = __builtin_amdgcn_wmma_f32_16x16x32_bf16(
          false, qa[1].bf, false, kb1.bf, (short)0, s, false, false);
      S[t] = s;
    }

    // ---- causal mask + row max (rows live in 16-lane halves) ----
    float rmax[8];
#pragma unroll
    for (int i = 0; i < 8; ++i) {
      int q = q0 + 8 * hi + i;
#pragma unroll
      for (int t = 0; t < 2; ++t) {
        float sv = S[t][i] * 0.125f;  // 64^-0.5
        if (j0 + 16 * t + r > q) sv = -1e30f;
        S[t][i] = sv;
      }
      rmax[i] = fmaxf(S[0][i], S[1][i]);
    }
#pragma unroll
    for (int i = 0; i < 8; ++i) {
#pragma unroll
      for (int off = 1; off < 16; off <<= 1)
        rmax[i] = fmaxf(rmax[i], __shfl_xor(rmax[i], off, 32));
    }

    // ---- online softmax update; P and row sums ----
    float scale_i[8], rsum[8];
#pragma unroll
    for (int i = 0; i < 8; ++i) {
      float mn = fmaxf(mrow[i], rmax[i]);
      scale_i[i] = __expf(mrow[i] - mn);
      float p0 = __expf(S[0][i] - mn);
      float p1 = __expf(S[1][i] - mn);
      S[0][i] = p0;
      S[1][i] = p1;
      rsum[i] = p0 + p1;
      mrow[i] = mn;
    }
#pragma unroll
    for (int i = 0; i < 8; ++i) {
#pragma unroll
      for (int off = 1; off < 16; off <<= 1)
        rsum[i] += __shfl_xor(rsum[i], off, 32);
      lrow[i] = lrow[i] * scale_i[i] + rsum[i];
    }
#pragma unroll
    for (int s = 0; s < 4; ++s)
#pragma unroll
      for (int i = 0; i < 8; ++i) oacc[s][i] *= scale_i[i];

    // ---- P: C layout -> A fragment via LDS transpose (bf16) ----
    __syncthreads();
#pragma unroll
    for (int t = 0; t < 2; ++t)
#pragma unroll
      for (int i = 0; i < 8; ++i)
        pbuf[(i + 8 * hi) * 32 + 16 * t + r] = f32_to_bf16(S[t][i]);
    __syncthreads();
    Frag pa;
    pa.u[0] = *(const u16x8*)(&pbuf[r * 32 + koffA]);
    pa.u[1] = *(const u16x8*)(&pbuf[r * 32 + koffA + 16]);

    // ---- PV: O(16x64) += P(16x32) @ V(32x64), 4 dim-subtiles ----
#pragma unroll
    for (int s = 0; s < 4; ++s) {
      const unsigned short* vrow = vbase + (size_t)(16 * s + r) * L_ + j0 + koffB;
      Frag vb;
      vb.u[0] = *(const u16x8*)(vrow);
      vb.u[1] = *(const u16x8*)(vrow + 16);
      oacc[s] = __builtin_amdgcn_wmma_f32_16x16x32_bf16(
          false, pa.bf, false, vb.bf, (short)0, oacc[s], false, false);
    }
  }

  // ---- finalize: 1/l and store bf16 ----
  float inv_i[8];
#pragma unroll
  for (int i = 0; i < 8; ++i) inv_i[i] = 1.0f / lrow[i];
#pragma unroll
  for (int s = 0; s < 4; ++s) {
#pragma unroll
    for (int i = 0; i < 8; ++i) {
      int m = q0 + 8 * hi + i;
      int d = 16 * s + r;
      out_bf[((size_t)b * L_ + m) * (NH_ * HD_) + h * HD_ + d] =
          f32_to_bf16(oacc[s][i] * inv_i[i]);
    }
  }
}

// ---------------- muP scale of in_proj output (in place) ----------------
__global__ void mup_scale_kernel(float* __restrict__ proj) {
  size_t n = (size_t)M_ * PROJ_;
  size_t i = (size_t)blockIdx.x * blockDim.x + threadIdx.x;
  size_t stride = (size_t)gridDim.x * blockDim.x;
  for (; i < n; i += stride) {
    int c = (int)(i % PROJ_);
    float m = (c < DSSM_)               ? 1.0f
              : (c < 2 * DSSM_)         ? 0.9f
              : (c < 2 * DSSM_ + MS_)   ? 0.8f
              : (c < 2 * DSSM_ + 2*MS_) ? 1.1f
                                        : 1.2f;
    proj[i] *= m;
  }
}

// ---------------- causal depthwise conv (MK taps) + SiLU ----------------
__global__ void conv_silu_kernel(const float* __restrict__ proj,
                                 const float* __restrict__ conv_w,
                                 const float* __restrict__ conv_b,
                                 float* __restrict__ xbc) {
  size_t n = (size_t)M_ * CONV_DIM_;
  size_t i = (size_t)blockIdx.x * blockDim.x + threadIdx.x;
  size_t stride = (size_t)gridDim.x * blockDim.x;
  for (; i < n; i += stride) {
    int c = (int)(i % CONV_DIM_);
    int row = (int)(i / CONV_DIM_);
    int l = row % L_;
    int b = row / L_;
    float s = conv_b[c];
#pragma unroll
    for (int kk = 0; kk < MK_; ++kk) {
      int li = l + kk - (MK_ - 1);
      if (li >= 0)
        s += proj[((size_t)b * L_ + li) * PROJ_ + DSSM_ + c] * conv_w[c * MK_ + kk];
    }
    xbc[i] = silu_f(s);
  }
}

// ---------------- selective-state scan: one block of 64 threads per (b,h) ----------
__global__ __launch_bounds__(64) void ssm_scan_kernel(
    const float* __restrict__ xbc, const float* __restrict__ proj,
    const float* __restrict__ A_log, const float* __restrict__ dt_bias,
    const float* __restrict__ Dvec, float* __restrict__ y) {
  int bh = blockIdx.x;
  int h = bh % MH_;
  int b = bh / MH_;
  int p = threadIdx.x;  // 0..63

  __shared__ float Bs[MS_];
  __shared__ float Cs[MS_];

  float state[MS_];
#pragma unroll
  for (int s = 0; s < MS_; ++s) state[s] = 0.0f;

  float A = -__expf(A_log[h]);
  float dtb = dt_bias[h];
  float Dh = Dvec[h];

  for (int l = 0; l < L_; ++l) {
    size_t row = (size_t)b * L_ + l;
    Bs[p] = xbc[row * CONV_DIM_ + DSSM_ + p];
    Cs[p] = xbc[row * CONV_DIM_ + DSSM_ + MS_ + p];
    float dtr = proj[row * PROJ_ + (PROJ_ - MH_) + h] + dtb;
    float dt = (dtr > 20.0f) ? dtr : log1pf(__expf(dtr));
    float dA = __expf(dt * A);
    float xp = xbc[row * CONV_DIM_ + h * MP_ + p];
    __syncthreads();
    float acc = 0.0f;
    float dtx = dt * xp;
#pragma unroll 8
    for (int s = 0; s < MS_; ++s) {
      state[s] = state[s] * dA + dtx * Bs[s];
      acc += state[s] * Cs[s];
    }
    y[(row * MH_ + h) * MP_ + p] = acc + xp * Dh;
    __syncthreads();
  }
}

// ---------------- g = y * silu(z); RMSNorm(g)*w -> bf16 ----------------
__global__ __launch_bounds__(256) void gate_rms_kernel(
    const float* __restrict__ y, const float* __restrict__ proj,
    const float* __restrict__ norm_w, unsigned short* __restrict__ g_bf) {
  __shared__ float red[8];
  int row = blockIdx.x;
  float loc[DSSM_ / 256];
  float ss = 0.0f;
#pragma unroll
  for (int j = 0; j < DSSM_ / 256; ++j) {
    int c = threadIdx.x + j * 256;
    float z = proj[(size_t)row * PROJ_ + c];  // muP factor for z is 1.0
    float g = y[(size_t)row * DSSM_ + c] * silu_f(z);
    loc[j] = g;
    ss += g * g;
  }
  float tot = blockReduceSum256(ss, red);
  float scale = rsqrtf(tot / DSSM_ + EPS_);
#pragma unroll
  for (int j = 0; j < DSSM_ / 256; ++j) {
    int c = threadIdx.x + j * 256;
    g_bf[(size_t)row * DSSM_ + c] = f32_to_bf16(loc[j] * scale * norm_w[c]);
  }
}

// ---------------- h = attn*0.8 + ssm*0.9 + resid; store h; RMSNorm(h)*w -> bf16 -----
__global__ __launch_bounds__(256) void combine_rms_kernel(
    const float* __restrict__ attn_h, const float* __restrict__ ssm_h,
    const float* __restrict__ resid, const float* __restrict__ w2,
    float* __restrict__ hmid, unsigned short* __restrict__ h2_bf) {
  __shared__ float red[8];
  int row = blockIdx.x;
  float loc[HID_ / 256];
  float ss = 0.0f;
#pragma unroll
  for (int j = 0; j < HID_ / 256; ++j) {
    size_t idx = (size_t)row * HID_ + threadIdx.x + j * 256;
    float h = attn_h[idx] * ATTN_OUT_ + ssm_h[idx] * SSM_OUT_ + resid[idx];
    loc[j] = h;
    hmid[idx] = h;
    ss += h * h;
  }
  float tot = blockReduceSum256(ss, red);
  float scale = rsqrtf(tot / HID_ + EPS_);
#pragma unroll
  for (int j = 0; j < HID_ / 256; ++j) {
    int c = threadIdx.x + j * 256;
    h2_bf[(size_t)row * HID_ + c] = f32_to_bf16(loc[j] * scale * w2[c]);
  }
}

// ---------------- SwiGLU activation: silu(gate*0.9)*up -> bf16 ----------------
__global__ void mlp_act_kernel(const float* __restrict__ gu,
                               unsigned short* __restrict__ act_bf) {
  size_t n = (size_t)M_ * INTER_;
  size_t i = (size_t)blockIdx.x * blockDim.x + threadIdx.x;
  size_t stride = (size_t)gridDim.x * blockDim.x;
  for (; i < n; i += stride) {
    int c = (int)(i % INTER_);
    size_t row = i / INTER_;
    float gate = gu[row * (2 * INTER_) + c] * GATE_MULT_;
    float up = gu[row * (2 * INTER_) + INTER_ + c];
    act_bf[i] = f32_to_bf16(silu_f(gate) * up);
  }
}

// ---------------- out = down*0.8 + residual2 ----------------
__global__ void final_add_kernel(const float* __restrict__ dn,
                                 const float* __restrict__ hmid,
                                 float* __restrict__ out) {
  size_t n = (size_t)M_ * HID_;
  size_t i = (size_t)blockIdx.x * blockDim.x + threadIdx.x;
  size_t stride = (size_t)gridDim.x * blockDim.x;
  for (; i < n; i += stride) out[i] = dn[i] * DOWN_MULT_ + hmid[i];
}

// ================= host side =================
static inline void launch_gemm(const unsigned short* A, const unsigned short* W,
                               float* C, int M, int N, int K, hipStream_t s) {
  dim3 grid((N + 255) / 256, M / 32);
  gemm_bf16_wmma<<<grid, 128, 0, s>>>(A, W, C, M, N, K);
}

extern "C" void kernel_launch(void* const* d_in, const int* in_sizes, int n_in,
                              void* d_out, int out_size, void* d_ws, size_t ws_size,
                              hipStream_t stream) {
  (void)in_sizes; (void)n_in; (void)out_size; (void)ws_size;
  const float* hidden   = (const float*)d_in[0];
  const int*   positions= (const int*)d_in[1];
  const float* w_in_ln  = (const float*)d_in[2];
  const float* qkv_w    = (const float*)d_in[3];
  const float* o_w      = (const float*)d_in[4];
  const float* in_proj_w= (const float*)d_in[5];
  const float* conv_w   = (const float*)d_in[6];
  const float* conv_b   = (const float*)d_in[7];
  const float* A_log    = (const float*)d_in[8];
  const float* dt_bias  = (const float*)d_in[9];
  const float* Dvec     = (const float*)d_in[10];
  const float* ssm_norm_w = (const float*)d_in[11];
  const float* out_proj_w = (const float*)d_in[12];
  const float* w_pre_ff = (const float*)d_in[13];
  const float* gate_up_w= (const float*)d_in[14];
  const float* down_w   = (const float*)d_in[15];
  float* out = (float*)d_out;

  // workspace bump allocator
  size_t off = 0;
  auto alloc = [&](size_t bytes) -> void* {
    void* p = (char*)d_ws + off;
    off = (off + bytes + 255) & ~(size_t)255;
    return p;
  };
  unsigned short* w_qkv_bf = (unsigned short*)alloc((size_t)QKV_ * HID_ * 2);
  unsigned short* w_o_bf   = (unsigned short*)alloc((size_t)HID_ * NH_ * HD_ * 2);
  unsigned short* w_in_bf  = (unsigned short*)alloc((size_t)PROJ_ * HID_ * 2);
  unsigned short* w_out_bf = (unsigned short*)alloc((size_t)HID_ * DSSM_ * 2);
  unsigned short* w_gu_bf  = (unsigned short*)alloc((size_t)(2 * INTER_) * HID_ * 2);
  unsigned short* w_dn_bf  = (unsigned short*)alloc((size_t)HID_ * INTER_ * 2);
  unsigned short* hs_attn  = (unsigned short*)alloc((size_t)M_ * HID_ * 2);
  unsigned short* hs_ssm   = (unsigned short*)alloc((size_t)M_ * HID_ * 2);
  float* qkv      = (float*)alloc((size_t)M_ * QKV_ * 4);
  unsigned short* k_bf  = (unsigned short*)alloc((size_t)M_ * NKV_ * HD_ * 2);
  unsigned short* vT_bf = (unsigned short*)alloc((size_t)M_ * NKV_ * HD_ * 2);
  unsigned short* attn_out_bf = (unsigned short*)alloc((size_t)M_ * NH_ * HD_ * 2);
  float* attn_h   = (float*)alloc((size_t)M_ * HID_ * 4);
  float* proj     = (float*)alloc((size_t)M_ * PROJ_ * 4);
  float* xbc      = (float*)alloc((size_t)M_ * CONV_DIM_ * 4);
  float* yssm     = (float*)alloc((size_t)M_ * DSSM_ * 4);
  unsigned short* g_bf = (unsigned short*)alloc((size_t)M_ * DSSM_ * 2);
  float* ssm_h    = (float*)alloc((size_t)M_ * HID_ * 4);
  float* hmid     = (float*)alloc((size_t)M_ * HID_ * 4);
  unsigned short* h2_bf = (unsigned short*)alloc((size_t)M_ * HID_ * 2);
  float* gu       = (float*)alloc((size_t)M_ * 2 * INTER_ * 4);
  unsigned short* act_bf = (unsigned short*)alloc((size_t)M_ * INTER_ * 2);
  float* dn_out   = (float*)alloc((size_t)M_ * HID_ * 4);

  auto cast = [&](const float* src, unsigned short* dst, size_t n) {
    int blocks = (int)((n + 255) / 256);
    cast_bf16_kernel<<<blocks, 256, 0, stream>>>(src, dst, n);
  };

  // 1) weight casts fp32 -> bf16
  cast(qkv_w,     w_qkv_bf, (size_t)QKV_ * HID_);
  cast(o_w,       w_o_bf,   (size_t)HID_ * NH_ * HD_);
  cast(in_proj_w, w_in_bf,  (size_t)PROJ_ * HID_);
  cast(out_proj_w,w_out_bf, (size_t)HID_ * DSSM_);
  cast(gate_up_w, w_gu_bf,  (size_t)(2 * INTER_) * HID_);
  cast(down_w,    w_dn_bf,  (size_t)HID_ * INTER_);

  // 2) input RMSNorm -> bf16 activations (x ATTN_IN, x SSM_IN)
  rmsnorm_in_kernel<<<M_, 256, 0, stream>>>(hidden, w_in_ln, hs_attn, hs_ssm);

  // 3) attention branch
  launch_gemm(hs_attn, w_qkv_bf, qkv, M_, QKV_, HID_, stream);
  {
    int total = M_ * (NH_ + NKV_) * 32;
    rope_kernel<<<(total + 255) / 256, 256, 0, stream>>>(qkv, positions);
  }
  {
    size_t n = (size_t)M_ * NKV_ * HD_;
    kv_extract_kernel<<<(int)((n + 255) / 256), 256, 0, stream>>>(qkv, k_bf, vT_bf);
  }
  attn_wmma_kernel<<<B_ * NH_ * (L_ / 16), 32, 0, stream>>>(qkv, k_bf, vT_bf, attn_out_bf);
  launch_gemm(attn_out_bf, w_o_bf, attn_h, M_, HID_, NH_ * HD_, stream);

  // 4) mamba branch
  launch_gemm(hs_ssm, w_in_bf, proj, M_, PROJ_, HID_, stream);
  {
    size_t n = (size_t)M_ * PROJ_;
    mup_scale_kernel<<<(int)((n + 255) / 256), 256, 0, stream>>>(proj);
  }
  {
    size_t n = (size_t)M_ * CONV_DIM_;
    conv_silu_kernel<<<(int)((n + 255) / 256), 256, 0, stream>>>(proj, conv_w, conv_b, xbc);
  }
  ssm_scan_kernel<<<B_ * MH_, 64, 0, stream>>>(xbc, proj, A_log, dt_bias, Dvec, yssm);
  gate_rms_kernel<<<M_, 256, 0, stream>>>(yssm, proj, ssm_norm_w, g_bf);
  launch_gemm(g_bf, w_out_bf, ssm_h, M_, HID_, DSSM_, stream);

  // 5) combine + pre-FF RMSNorm
  combine_rms_kernel<<<M_, 256, 0, stream>>>(attn_h, ssm_h, hidden, w_pre_ff, hmid, h2_bf);

  // 6) MLP
  launch_gemm(h2_bf, w_gu_bf, gu, M_, 2 * INTER_, HID_, stream);
  {
    size_t n = (size_t)M_ * INTER_;
    mlp_act_kernel<<<(int)((n + 255) / 256), 256, 0, stream>>>(gu, act_bf);
  }
  launch_gemm(act_bf, w_dn_bf, dn_out, M_, HID_, INTER_, stream);

  // 7) final residual
  {
    size_t n = (size_t)M_ * HID_;
    final_add_kernel<<<(int)((n + 255) / 256), 256, 0, stream>>>(dn_out, hmid, out);
  }
}
